// GPSNet_39402029973508
// MI455X (gfx1250) — matrix operand order repaired and checked by reading, not compiled, not run on hardware
//
#include <hip/hip_runtime.h>
#include <cstdint>
#include <cstddef>

// ---------------- problem constants (match reference) ----------------
#define NNODES 16384
#define NEDGES 262144
#define HID    128
#define NB     32
#define NS     512
#define NHEADS 4
#define DH     32
#define OUTD   64

typedef __attribute__((ext_vector_type(16))) _Float16 v16h;
typedef __attribute__((ext_vector_type(8)))  _Float16 v8h;
typedef __attribute__((ext_vector_type(8)))  float    v8f;
typedef __attribute__((ext_vector_type(4)))  unsigned int v4u;
typedef __attribute__((ext_vector_type(8)))  int v8i;
typedef __attribute__((ext_vector_type(4)))  int v4i;

union V16H { v16h v; v8h h[2]; };

#if defined(__has_builtin)
#if __has_builtin(__builtin_amdgcn_tensor_load_to_lds) && __has_builtin(__builtin_amdgcn_s_wait_tensorcnt)
#define USE_TDM 1
#endif
#endif
#ifndef USE_TDM
#define USE_TDM 0
#endif

// ---------------- Tensor Data Mover: 2-D tile (rows x rowlen f16) -> LDS ----------------
// D# layout per CDNA5 ISA (group0 128b, group1 256b; groups 2/3 unused for 2D).
#if USE_TDM
__device__ __forceinline__ void tdm_load_2d_f16(unsigned lds_off, const void* gaddr,
                                                unsigned rows, unsigned rowlen,
                                                unsigned stride) {
  unsigned long long ga = (unsigned long long)(uintptr_t)gaddr;
  v4u g0;
  g0.x = 1u;                                          // count=1, user-mode descriptor
  g0.y = lds_off;                                     // lds_addr (bytes)
  g0.z = (unsigned)(ga & 0xffffffffu);                // global_addr[31:0]
  g0.w = (unsigned)((ga >> 32) & 0x01ffffffu)         // global_addr[56:32]
         | (2u << 30);                                // type=2 ("image")
  v8i g1;
  g1[0] = (int)(1u << 16);                            // wg_mask=0, data_size=1 (2 bytes)
  g1[1] = (int)((rowlen & 0xffffu) << 16);            // tensor_dim0[15:0]   (bits 63:48)
  g1[2] = (int)((rowlen >> 16) & 0xffffu)             // tensor_dim0[31:16]
          | (int)((rows & 0xffffu) << 16);            // tensor_dim1[15:0]
  g1[3] = (int)((rows >> 16) & 0xffffu)               // tensor_dim1[31:16]
          | (int)((rowlen & 0xffffu) << 16);          // tile_dim0
  g1[4] = (int)(rows & 0xffffu);                      // tile_dim1 (tile_dim2 = 0)
  g1[5] = (int)stride;                                // tensor_dim0_stride[31:0]
  g1[6] = 0;                                          // stride hi / dim1_stride lo
  g1[7] = 0;
  v4i z = {0, 0, 0, 0};
#if defined(__clang_major__) && __clang_major__ >= 23
  v8i z8 = {0, 0, 0, 0, 0, 0, 0, 0};
  __builtin_amdgcn_tensor_load_to_lds(g0, g1, z, z, z8, 0);
#else
  __builtin_amdgcn_tensor_load_to_lds(g0, g1, z, z, 0);
#endif
}
#endif

// ---------------- WMMA fragment loaders (wave32, 16x16x32 f16) ----------------
__device__ __forceinline__ v16h load_a_frag(const _Float16* __restrict__ A, int lda,
                                            int m0, int k0, int lane) {
  const int lg = lane >> 4, lr = lane & 15;
  const _Float16* row = A + (size_t)(m0 + lr) * lda + k0 + 8 * lg;
  V16H u;
  u.h[0] = *(const v8h*)(row);
  u.h[1] = *(const v8h*)(row + 16);
  return u.v;
}

__device__ __forceinline__ v16h load_b_frag(const _Float16* __restrict__ W, int ldw,
                                            int n0, int k0, int lane) {
  const int lg = lane >> 4, lr = lane & 15;
  const _Float16* row = W + (size_t)(n0 + lr) * ldw + k0 + 8 * lg;
  V16H u;
  u.h[0] = *(const v8h*)(row);
  u.h[1] = *(const v8h*)(row + 16);
  return u.v;
}

// ---------------- fused WMMA GEMM:  out = epilogue(A @ W^T [+ A2 @ W2^T]) ----------------
// 4 waves / block; each wave owns one 16-row m-tile; the 16-col weight panel(s) for the
// block's n-tile are staged once into LDS by the TDM (wave 0), then consumed via ds_load_b128.
enum EpMode { EP_BIAS, EP_BIAS_BN_RELU, EP_BIAS_RES_BN, EP_OUTPROJ, EP_BIAS_RELU, EP_BIAS_F32 };

#define PANEL_HALFS (16 * 256)   // max K = 256

template<EpMode MODE, bool DUAL>
__global__ __launch_bounds__(128)
void wmma_gemm_kernel(const _Float16* __restrict__ A, const _Float16* __restrict__ W,
                      const _Float16* __restrict__ A2, const _Float16* __restrict__ W2,
                      int K, int lda, int ldw, int ldo,
                      const float* __restrict__ bias,
                      const float* __restrict__ bnsc, const float* __restrict__ bnsh,
                      const _Float16* __restrict__ res, const _Float16* __restrict__ h1,
                      _Float16* __restrict__ out16, float* __restrict__ out32)
{
  __shared__ __attribute__((aligned(16))) _Float16 wpan[2 * PANEL_HALFS];

  const int wave = threadIdx.x >> 5;
  const int lane = threadIdx.x & 31;
  const int m0 = (blockIdx.x * 4 + wave) * 16;
  const int n0 = blockIdx.y * 16;

#if USE_TDM
  if (wave == 0) {
    tdm_load_2d_f16(0, W + (size_t)n0 * ldw, 16u, (unsigned)K, (unsigned)ldw);
    if constexpr (DUAL)
      tdm_load_2d_f16(PANEL_HALFS * 2, W2 + (size_t)n0 * ldw, 16u, (unsigned)K, (unsigned)ldw);
    __builtin_amdgcn_s_wait_tensorcnt(0);
  }
#else
  for (int idx = threadIdx.x; idx < 16 * K; idx += 128)
    wpan[idx] = W[(size_t)(n0 + idx / K) * ldw + (idx % K)];
  if constexpr (DUAL)
    for (int idx = threadIdx.x; idx < 16 * K; idx += 128)
      wpan[PANEL_HALFS + idx] = W2[(size_t)(n0 + idx / K) * ldw + (idx % K)];
#endif
  __syncthreads();

  v8f acc = {};
  for (int k0 = 0; k0 < K; k0 += 32) {
    v16h a = load_a_frag(A, lda, m0, k0, lane);
    v16h b = load_b_frag(wpan, K, 0, k0, lane);
    acc = __builtin_amdgcn_wmma_f32_16x16x32_f16(false, a, false, b, (short)0, acc, false, false);
    if constexpr (DUAL) {
      v16h a2 = load_a_frag(A2, lda, m0, k0, lane);
      v16h b2 = load_b_frag(wpan + PANEL_HALFS, K, 0, k0, lane);
      acc = __builtin_amdgcn_wmma_f32_16x16x32_f16(false, a2, false, b2, (short)0, acc, false, false);
    }
  }

  const int lg = lane >> 4, lr = lane & 15;
  const int n = n0 + lr;
  const float bv = bias[n];
  float s = 1.f, t = 0.f;
  if constexpr (MODE == EP_BIAS_BN_RELU || MODE == EP_BIAS_RES_BN || MODE == EP_OUTPROJ) {
    s = bnsc[n]; t = bnsh[n];
  }
#pragma unroll
  for (int i = 0; i < 8; ++i) {
    const int m = m0 + i + 8 * lg;
    float v = acc[i] + bv;
    if constexpr (MODE == EP_BIAS_RES_BN || MODE == EP_OUTPROJ)
      v += (float)res[(size_t)m * HID + n];
    if constexpr (MODE == EP_BIAS_BN_RELU || MODE == EP_BIAS_RES_BN || MODE == EP_OUTPROJ)
      v = v * s + t;
    if constexpr (MODE == EP_OUTPROJ)
      v += (float)h1[(size_t)m * HID + n];
    if constexpr (MODE == EP_BIAS_BN_RELU || MODE == EP_BIAS_RELU)
      v = fmaxf(v, 0.f);
    if constexpr (MODE == EP_BIAS_F32)
      out32[(size_t)m * ldo + n] = v;
    else
      out16[(size_t)m * ldo + n] = (_Float16)v;
  }
}

// ---------------- attention: one wave per (16-row q tile, batch*head) ----------------
// dyn LDS: sc[16][512] f32 (reused as vt[32][512] f16)  |  at[16][512] f16  |  rs[16] f32
#define ATT_SMEM (16 * NS * 4 + 16 * NS * 2 + 64)

__global__ __launch_bounds__(32)
void attention_kernel(const _Float16* __restrict__ qkv, _Float16* __restrict__ o16)
{
  extern __shared__ char smem[];
  float*    sc = (float*)smem;                          // [16][512] scores (f32)
  _Float16* vt = (_Float16*)smem;                       // [32][512] V^T (overlaps sc)
  _Float16* at = (_Float16*)(smem + 16 * NS * 4);       // [16][512] exp(scores) (f16)
  float*    rs = (float*)(smem + 16 * NS * 4 + 16 * NS * 2); // [16] row sums

  const int qt   = blockIdx.x;            // 0..31  (16-row q tile)
  const int bh   = blockIdx.y;            // 0..127
  const int b    = bh / NHEADS;
  const int h    = bh % NHEADS;
  const int lane = threadIdx.x;
  const int lg = lane >> 4, lr = lane & 15;

  const _Float16* base = qkv + (size_t)b * NS * (3 * HID);
  const _Float16* qm = base + h * DH;                   // q[i][d] @ qm + i*384 + d
  const _Float16* km = base + HID + h * DH;
  const _Float16* vm = base + 2 * HID + h * DH;

  const int i0 = qt * 16;
  const float scl = 0.17677669529663687f;               // 1/sqrt(32)

  // ---- scores = q @ k^T / sqrt(dh) : K=32=dh, one WMMA per 16x16 tile ----
  v16h aq = load_a_frag(qm, 3 * HID, i0, 0, lane);
  for (int jt = 0; jt < NS / 16; ++jt) {
    v16h bk = load_b_frag(km, 3 * HID, jt * 16, 0, lane);
    v8f c = {};
    c = __builtin_amdgcn_wmma_f32_16x16x32_f16(false, aq, false, bk, (short)0, c, false, false);
#pragma unroll
    for (int i = 0; i < 8; ++i)
      sc[(size_t)(i + 8 * lg) * NS + jt * 16 + lr] = c[i] * scl;
  }
  __syncthreads();

  // ---- row softmax (16 rows, whole wave per row) ----
  for (int r = 0; r < 16; ++r) {
    float mx = -1e30f;
    for (int cc = lane; cc < NS; cc += 32) mx = fmaxf(mx, sc[(size_t)r * NS + cc]);
    for (int off = 16; off > 0; off >>= 1) mx = fmaxf(mx, __shfl_xor(mx, off, 32));
    float sum = 0.f;
    for (int cc = lane; cc < NS; cc += 32) {
      float e = __expf(sc[(size_t)r * NS + cc] - mx);
      at[(size_t)r * NS + cc] = (_Float16)e;
      sum += e;
    }
    for (int off = 16; off > 0; off >>= 1) sum += __shfl_xor(sum, off, 32);
    if (lane == 0) rs[r] = sum;
  }
  __syncthreads();

  // ---- stage V transposed into LDS (overwrites score region) ----
  for (int idx = lane; idx < NS * DH; idx += 32) {
    int j = idx >> 5, d = idx & 31;
    vt[(size_t)d * NS + j] = vm[(size_t)j * (3 * HID) + d];
  }
  __syncthreads();

  // ---- o = softmax(scores) @ V : K-loop over S in steps of 32 ----
  _Float16* orow = o16 + ((size_t)(b * NS + i0)) * HID + h * DH;
  for (int dt = 0; dt < 2; ++dt) {
    v8f acc = {};
    for (int k0 = 0; k0 < NS; k0 += 32) {
      v16h a  = load_a_frag(at, NS, 0, k0, lane);
      v16h bv = load_b_frag(vt, NS, dt * 16, k0, lane);
      acc = __builtin_amdgcn_wmma_f32_16x16x32_f16(false, a, false, bv, (short)0, acc, false, false);
    }
#pragma unroll
    for (int i = 0; i < 8; ++i) {
      const int m = i + 8 * lg;
      orow[(size_t)m * HID + dt * 16 + lr] = (_Float16)(acc[i] / rs[m]);
    }
  }
}

// ---------------- small utility kernels ----------------
__global__ void f32_to_f16_kernel(const float* __restrict__ in, _Float16* __restrict__ out, size_t n) {
  for (size_t i = (size_t)blockIdx.x * blockDim.x + threadIdx.x; i < n;
       i += (size_t)gridDim.x * blockDim.x)
    out[i] = (_Float16)in[i];
}

__global__ void zero_f32_kernel(float* __restrict__ p, size_t n) {
  for (size_t i = (size_t)blockIdx.x * blockDim.x + threadIdx.x; i < n;
       i += (size_t)gridDim.x * blockDim.x)
    p[i] = 0.f;
}

__global__ void bnfold_kernel(const float* __restrict__ g, const float* __restrict__ beta,
                              const float* __restrict__ rm, const float* __restrict__ rv,
                              float* __restrict__ sc, float* __restrict__ sh, int c) {
  int i = blockIdx.x * blockDim.x + threadIdx.x;
  if (i < c) {
    float s = g[i] * rsqrtf(rv[i] + 1e-5f);
    sc[i] = s;
    sh[i] = beta[i] - rm[i] * s;
  }
}

// one wave per edge: gather x[src] (f16), atomic-add into agg[dst] (f32), count edges
__global__ void scatter_add_kernel(const _Float16* __restrict__ x, const int* __restrict__ src,
                                   const int* __restrict__ dst, float* __restrict__ aggs,
                                   float* __restrict__ cnt) {
  const int e = blockIdx.x * (blockDim.x >> 5) + (threadIdx.x >> 5);
  const int lane = threadIdx.x & 31;
  if (e >= NEDGES) return;
  const int s = src[e], d = dst[e];
  const _Float16* xs = x + (size_t)s * HID;
  float* ad = aggs + (size_t)d * HID;
#pragma unroll
  for (int i = 0; i < HID / 32; ++i) {
    int c = lane + 32 * i;
    atomicAdd(ad + c, (float)xs[c]);
  }
  if (lane == 0) atomicAdd(cnt + d, 1.0f);
}

__global__ void agg_norm_kernel(const float* __restrict__ s, const float* __restrict__ cnt,
                                _Float16* __restrict__ out, size_t n) {
  for (size_t i = (size_t)blockIdx.x * blockDim.x + threadIdx.x; i < n;
       i += (size_t)gridDim.x * blockDim.x) {
    float c = cnt[i >> 7];
    c = c < 1.f ? 1.f : c;
    out[i] = (_Float16)(s[i] / c);
  }
}

__global__ void bnrelu_f16_kernel(const _Float16* __restrict__ in, const float* __restrict__ sc,
                                  const float* __restrict__ sh, _Float16* __restrict__ out, size_t n) {
  for (size_t i = (size_t)blockIdx.x * blockDim.x + threadIdx.x; i < n;
       i += (size_t)gridDim.x * blockDim.x) {
    int c = (int)(i & (HID - 1));
    float v = (float)in[i] * sc[c] + sh[c];
    out[i] = (_Float16)fmaxf(v, 0.f);
  }
}

// ---------------- host-side orchestration ----------------
struct BNp { float* sc; float* sh; };

extern "C" void kernel_launch(void* const* d_in, const int* in_sizes, int n_in,
                              void* d_out, int out_size, void* d_ws, size_t ws_size,
                              hipStream_t stream)
{
  (void)in_sizes; (void)n_in; (void)out_size; (void)ws_size;
  const float* xf = (const float*)d_in[0];
  const int* ei = (const int*)d_in[1];
  const int* srcp = ei;
  const int* dstp = ei + NEDGES;

  auto F = [&](int i) { return (const float*)d_in[i]; };

  // flattened parameter indices (setup_inputs dict order, depth-first)
  const int P_LIN_IN_W = 3, P_LIN_IN_B = 4, P_BN1 = 5, G1 = 9, P_BN2 = 32, G2 = 36,
            P_LOUT_W = 59, P_LOUT_B = 60;
  const int O_SL_W = 0, O_SL_B = 1, O_SR_W = 2, O_N1 = 3, O_AI_W = 7, O_AI_B = 8,
            O_AO_W = 9, O_AO_B = 10, O_N2 = 11, O_M1_W = 15, O_M1_B = 16,
            O_M2_W = 17, O_M2_B = 18, O_N3 = 19;

  char* wsp = (char*)d_ws;
  auto alloc = [&](size_t bytes) -> char* {
    char* p = wsp; wsp += (bytes + 255) & ~(size_t)255; return p;
  };

  _Float16* x16   = (_Float16*)alloc((size_t)NNODES * HID * 2);
  _Float16* h0    = (_Float16*)alloc((size_t)NNODES * HID * 2);
  _Float16* hA    = (_Float16*)alloc((size_t)NNODES * HID * 2);
  _Float16* hB    = (_Float16*)alloc((size_t)NNODES * HID * 2);
  _Float16* hC    = (_Float16*)alloc((size_t)NNODES * HID * 2);
  float*    agg32 = (float*)   alloc((size_t)NNODES * HID * 4);
  float*    cnt   = (float*)   alloc((size_t)NNODES * 4);
  _Float16* agg16 = (_Float16*)alloc((size_t)NNODES * HID * 2);
  _Float16* h1b   = (_Float16*)alloc((size_t)NNODES * HID * 2);
  _Float16* qkv   = (_Float16*)alloc((size_t)NNODES * 3 * HID * 2);
  _Float16* obuf  = (_Float16*)alloc((size_t)NNODES * HID * 2);
  _Float16* outb  = (_Float16*)alloc((size_t)NNODES * HID * 2);
  _Float16* tbuf  = (_Float16*)alloc((size_t)NNODES * 2 * HID * 2);

  auto cvt = [&](int idx, size_t nelem) -> _Float16* {
    _Float16* p = (_Float16*)alloc(nelem * 2);
    f32_to_f16_kernel<<<dim3(256), dim3(256), 0, stream>>>(F(idx), p, nelem);
    return p;
  };
  auto bnf = [&](int gidx) -> BNp {
    BNp b;
    b.sc = (float*)alloc(HID * 4);
    b.sh = (float*)alloc(HID * 4);
    bnfold_kernel<<<dim3(1), dim3(HID), 0, stream>>>(F(gidx), F(gidx + 1), F(gidx + 2),
                                                     F(gidx + 3), b.sc, b.sh, HID);
    return b;
  };

  _Float16* w_lin_in = cvt(P_LIN_IN_W, 128 * 128);
  BNp bn1 = bnf(P_BN1);
  BNp bn2 = bnf(P_BN2);
  _Float16* w_lout = cvt(P_LOUT_W, (size_t)OUTD * 128);

  struct GW {
    _Float16 *wl, *wr, *wai, *wao, *wm1, *wm2;
    BNp n1, n2, n3;
    const float *bl, *bai, *bao, *bm1, *bm2;
  };
  auto prep = [&](int g) -> GW {
    GW o;
    o.wl  = cvt(g + O_SL_W, 128 * 128);
    o.wr  = cvt(g + O_SR_W, 128 * 128);
    o.wai = cvt(g + O_AI_W, 384 * 128);
    o.wao = cvt(g + O_AO_W, 128 * 128);
    o.wm1 = cvt(g + O_M1_W, 256 * 128);
    o.wm2 = cvt(g + O_M2_W, 128 * 256);
    o.n1 = bnf(g + O_N1); o.n2 = bnf(g + O_N2); o.n3 = bnf(g + O_N3);
    o.bl = F(g + O_SL_B); o.bai = F(g + O_AI_B); o.bao = F(g + O_AO_B);
    o.bm1 = F(g + O_M1_B); o.bm2 = F(g + O_M2_B);
    return o;
  };
  GW g1 = prep(G1), g2 = prep(G2);

  // x -> f16
  f32_to_f16_kernel<<<dim3(1024), dim3(256), 0, stream>>>(xf, x16, (size_t)NNODES * HID);

  // h0 = relu(bn1(lin_in(x)))
  wmma_gemm_kernel<EP_BIAS_BN_RELU, false><<<dim3(NNODES / 64, HID / 16), 128, 0, stream>>>(
      x16, w_lin_in, nullptr, nullptr, 128, HID, HID, HID,
      F(P_LIN_IN_B), bn1.sc, bn1.sh, nullptr, nullptr, h0, nullptr);

  auto gps = [&](const _Float16* xin, const GW& g, _Float16* hout) {
    // scatter-mean aggregation
    zero_f32_kernel<<<dim3(2048), dim3(256), 0, stream>>>(agg32, (size_t)NNODES * HID);
    zero_f32_kernel<<<dim3(64), dim3(256), 0, stream>>>(cnt, (size_t)NNODES);
    scatter_add_kernel<<<dim3(NEDGES / 8), dim3(256), 0, stream>>>(xin, srcp, dstp, agg32, cnt);
    agg_norm_kernel<<<dim3(2048), dim3(256), 0, stream>>>(agg32, cnt, agg16, (size_t)NNODES * HID);

    // h1 = bn_n1(lin_l(agg) + x @ Wr^T + x)     (dual-A WMMA)
    wmma_gemm_kernel<EP_BIAS_RES_BN, true><<<dim3(NNODES / 64, HID / 16), 128, 0, stream>>>(
        agg16, g.wl, xin, g.wr, 128, HID, HID, HID,
        g.bl, g.n1.sc, g.n1.sh, xin, nullptr, h1b, nullptr);

    // qkv = x @ Win^T + b
    wmma_gemm_kernel<EP_BIAS, false><<<dim3(NNODES / 64, 384 / 16), 128, 0, stream>>>(
        xin, g.wai, nullptr, nullptr, 128, HID, HID, 3 * HID,
        g.bai, nullptr, nullptr, nullptr, nullptr, qkv, nullptr);

    // per-(batch,head) attention
    attention_kernel<<<dim3(NS / 16, NB * NHEADS), 32, ATT_SMEM, stream>>>(qkv, obuf);

    // out = h1 + bn_n2(out_proj(o) + b + x)
    wmma_gemm_kernel<EP_OUTPROJ, false><<<dim3(NNODES / 64, HID / 16), 128, 0, stream>>>(
        obuf, g.wao, nullptr, nullptr, 128, HID, HID, HID,
        g.bao, g.n2.sc, g.n2.sh, xin, h1b, outb, nullptr);

    // t = relu(mlp1(out))
    wmma_gemm_kernel<EP_BIAS_RELU, false><<<dim3(NNODES / 64, 256 / 16), 128, 0, stream>>>(
        outb, g.wm1, nullptr, nullptr, 128, HID, HID, 2 * HID,
        g.bm1, nullptr, nullptr, nullptr, nullptr, tbuf, nullptr);

    // h = bn_n3(out + mlp2(t))
    wmma_gemm_kernel<EP_BIAS_RES_BN, false><<<dim3(NNODES / 64, HID / 16), 128, 0, stream>>>(
        tbuf, g.wm2, nullptr, nullptr, 256, 2 * HID, 2 * HID, HID,
        g.bm2, g.n3.sc, g.n3.sh, outb, nullptr, hout, nullptr);
  };

  gps(h0, g1, hA);
  bnrelu_f16_kernel<<<dim3(2048), dim3(256), 0, stream>>>(hA, bn2.sc, bn2.sh, hB,
                                                          (size_t)NNODES * HID);
  gps(hB, g2, hC);

  // final projection to f32 output [N, 64]
  wmma_gemm_kernel<EP_BIAS_F32, false><<<dim3(NNODES / 64, OUTD / 16), 128, 0, stream>>>(
      hC, w_lout, nullptr, nullptr, 128, HID, HID, OUTD,
      F(P_LOUT_B), nullptr, nullptr, nullptr, nullptr, nullptr, (float*)d_out);
}
